// MultiHeadScaledProductAttention_23184233464740
// MI455X (gfx1250) — matrix operand order repaired and checked by reading, not compiled, not run on hardware
//
#include <hip/hip_runtime.h>
#include <hip/hip_bf16.h>

typedef __attribute__((ext_vector_type(16))) __bf16 v16bf;
typedef __attribute__((ext_vector_type(2)))  __bf16 v2bf;
typedef __attribute__((ext_vector_type(8)))  float  v8f;

#define B_    16
#define TQ_   1024
#define TM_   1024
#define H_    8
#define NEGV  (-4294967295.0f)   // -(2^32)+1

// ---------- WMMA fragment loaders (16x16x32 bf16, contiguous k order used by both A and B) ----------
// A (16x32, M x K): lane L holds row M = L%16; VGPR t (pairs): K = (L/16)*16 + t*2 {+0,+1}
__device__ __forceinline__ v16bf load_a16x32(const __bf16* base, int stride, int lane) {
  const int m = lane & 15, half = lane >> 4;
  v16bf a;
#pragma unroll
  for (int t = 0; t < 8; ++t) {
    const int kk = (half << 4) + (t << 1);
    v2bf p = *(const v2bf*)(base + (size_t)m * stride + kk);
    a[2 * t]     = p.x;
    a[2 * t + 1] = p.y;
  }
  return a;
}

// B (32x16, K x N) sourced from an N-major (transposed) buffer so each lane's
// 16 K-elements are contiguous: lane L holds column N = L%16, K = (L/16)*16 + j.
__device__ __forceinline__ v16bf load_b32x16(const __bf16* base, int stride, int lane) {
  const int n = lane & 15, kb = (lane >> 4) << 4;
  return *(const v16bf*)(base + (size_t)n * stride + kb);
}

// ---------- fp32 -> bf16 conversion ----------
__global__ void cvt_act(const float* __restrict__ in0, const float* __restrict__ in1,
                        __bf16* __restrict__ o0, __bf16* __restrict__ o1, int n_each) {
  int i = blockIdx.x * blockDim.x + threadIdx.x;
  const int stride = gridDim.x * blockDim.x;
  for (; i < n_each; i += stride) {
    o0[i] = (__bf16)in0[i];
    o1[i] = (__bf16)in1[i];
  }
}

// W [512(k) x 512(n)] row-major -> WT [512(n) x 512(k)] bf16 (so B-fragments are contiguous)
__global__ void cvt_wT(const float* __restrict__ wq, const float* __restrict__ wk,
                       const float* __restrict__ wv,
                       __bf16* __restrict__ wqT, __bf16* __restrict__ wkT, __bf16* __restrict__ wvT) {
  const int i = blockIdx.x * blockDim.x + threadIdx.x;   // 0 .. 512*512-1
  const int k = i >> 9, n = i & 511;
  const int o = n * 512 + k;
  wqT[o] = (__bf16)wq[i];
  wkT[o] = (__bf16)wk[i];
  wvT[o] = (__bf16)wv[i];
}

// ---------- projection GEMM: Y[16384 x 512] = X[16384 x 512] @ W (via WT) ----------
// one wave per 16x16 output tile, K=512 -> 16 WMMAs
__global__ __launch_bounds__(32) void proj_gemm(const __bf16* __restrict__ X,
                                                const __bf16* __restrict__ WT,
                                                __bf16* __restrict__ Y,
                                                int transposed_store) {
  const int lane = threadIdx.x;
  const int m0 = blockIdx.x * 16;
  const int n0 = blockIdx.y * 16;
  v8f c = {};
#pragma unroll 4
  for (int k0 = 0; k0 < 512; k0 += 32) {
    v16bf a = load_a16x32(X + (size_t)m0 * 512 + k0, 512, lane);
    v16bf b = load_b32x16(WT + (size_t)n0 * 512 + k0, 512, lane);
    c = __builtin_amdgcn_wmma_f32_16x16x32_bf16(false, a, false, b, (short)0, c, false, false);
  }
  const int half = lane >> 4, nn = n0 + (lane & 15);
#pragma unroll
  for (int r = 0; r < 8; ++r) {
    const int m = m0 + r + (half << 3);            // global row (b*T + t)
    if (!transposed_store) {
      Y[(size_t)m * 512 + nn] = (__bf16)c[r];      // Q/K: [B*T, 512]
    } else {
      const int bb = m >> 10, t = m & 1023;        // V: store as Vt[b][d][t]
      Y[((size_t)bb * 512 + nn) * 1024 + t] = (__bf16)c[r];
    }
  }
}

// ---------- fused attention: one wave per (b, h, 16 q-rows) ----------
__global__ __launch_bounds__(32) void attn_kernel(const __bf16* __restrict__ Qb,
                                                  const __bf16* __restrict__ Kb,
                                                  const __bf16* __restrict__ Vt,
                                                  const int* __restrict__ mem_len,
                                                  const int* __restrict__ qry_len,
                                                  const int* __restrict__ caus,
                                                  float* __restrict__ ctx,
                                                  float* __restrict__ ali) {
  __shared__ float  lg[16 * 1024];                 // 64 KB: f32 logits, later bf16 probs in-place
  __shared__ __bf16 qlds[16 * 64];                 // 2 KB: Q tile staged via async-to-LDS
  const int lane = threadIdx.x;
  const int q0 = blockIdx.x * 16;
  const int h  = blockIdx.y;
  const int b  = blockIdx.z;

  const __bf16* Qbase = Qb + ((size_t)b * TQ_ + q0) * 512 + h * 64;
  const __bf16* Kbase = Kb + (size_t)b * TM_ * 512 + h * 64;

  // ---- Async-stage the 16x64 Q tile into LDS (overlaps the scalar length loads) ----
  // 2048 bytes = 4 wave-instructions x 32 lanes x 16B. ASYNCcnt-tracked.
  {
    const unsigned ldsbase = (unsigned)(uintptr_t)&qlds[0];
#pragma unroll
    for (int it = 0; it < 4; ++it) {
      const int idx   = it * 32 + lane;            // 0..127 16B-chunks
      const int row   = idx >> 3;                  // q row 0..15
      const int chunk = idx & 7;                   // 16B chunk within row
      const char*   g = (const char*)(Qbase + (size_t)row * 512 + chunk * 8);
      const unsigned l = ldsbase + (unsigned)(row * 128 + chunk * 16);
      asm volatile("global_load_async_to_lds_b128 %0, %1, off"
                   :: "v"(l), "v"(g) : "memory");
    }
  }

  const int ml = mem_len[b], ql = qry_len[b], cz = caus[0];
  const int half = lane >> 4, lmod = lane & 15;

  asm volatile("s_wait_asynccnt 0x0" ::: "memory");

  // ---- Phase 1: logits = (Q Kt)/8, masked, into LDS (128 WMMAs) ----
  for (int nt = 0; nt < TM_ / 16; ++nt) {
    const int n0 = nt * 16;
    v8f c = {};
#pragma unroll
    for (int k0 = 0; k0 < 64; k0 += 32) {
      v16bf a  = load_a16x32((const __bf16*)qlds + k0, 64, lane);   // loop-invariant, LDS-resident
      v16bf bf = load_b32x16(Kbase + (size_t)n0 * 512 + k0, 512, lane);
      c = __builtin_amdgcn_wmma_f32_16x16x32_bf16(false, a, false, bf, (short)0, c, false, false);
    }
    const int t = n0 + lmod;
#pragma unroll
    for (int r = 0; r < 8; ++r) {
      const int mrow = r + (half << 3);
      const int q = q0 + mrow;
      const bool ok = (t < ml) && (q < ql) && (!cz || (t <= q));
      lg[mrow * 1024 + t] = ok ? c[r] * 0.125f : NEGV;
    }
  }

  // ---- Phase 2: per-row softmax; write f32 alignments; pack bf16 probs in-place ----
  __bf16* P = (__bf16*)lg;
  for (int m = 0; m < 16; ++m) {
    float vals[32];
#pragma unroll
    for (int i = 0; i < 32; ++i) vals[i] = lg[m * 1024 + lane + 32 * i];  // read row fully first
    float mx = vals[0];
#pragma unroll
    for (int i = 1; i < 32; ++i) mx = fmaxf(mx, vals[i]);
#pragma unroll
    for (int off = 16; off > 0; off >>= 1) mx = fmaxf(mx, __shfl_xor(mx, off, 32));
    float s = 0.f;
#pragma unroll
    for (int i = 0; i < 32; ++i) { vals[i] = __expf(vals[i] - mx); s += vals[i]; }
#pragma unroll
    for (int off = 16; off > 0; off >>= 1) s += __shfl_xor(s, off, 32);
    const float inv = 1.0f / s;
    const int q = q0 + m;
    float*  arow = ali + (((size_t)(b * H_ + h) * TQ_) + q) * TM_;
    __bf16* prow = P + m * 1024;                   // bytes [2048m,2048m+2048): never clobbers unread f32
#pragma unroll
    for (int i = 0; i < 32; ++i) {
      const float p = vals[i] * inv;
      arow[lane + 32 * i] = p;
      prow[lane + 32 * i] = (__bf16)p;
    }
  }

  // ---- Phase 3: contexts = P @ V (128 WMMAs, A from LDS, B from transposed V) ----
  const __bf16* Vbase = Vt + ((size_t)b * 512 + h * 64) * 1024;
#pragma unroll
  for (int nt = 0; nt < 4; ++nt) {
    const int n0 = nt * 16;
    v8f c = {};
    for (int k0 = 0; k0 < TM_; k0 += 32) {
      v16bf a  = load_a16x32((const __bf16*)P + k0, 1024, lane);
      v16bf bf = load_b32x16(Vbase + (size_t)n0 * 1024 + k0, 1024, lane);
      c = __builtin_amdgcn_wmma_f32_16x16x32_bf16(false, a, false, bf, (short)0, c, false, false);
    }
#pragma unroll
    for (int r = 0; r < 8; ++r) {
      const int q = q0 + r + (half << 3);
      ctx[((size_t)b * TQ_ + q) * 512 + h * 64 + n0 + lmod] = c[r];
    }
  }
}

extern "C" void kernel_launch(void* const* d_in, const int* in_sizes, int n_in,
                              void* d_out, int out_size, void* d_ws, size_t ws_size,
                              hipStream_t stream) {
  (void)in_sizes; (void)n_in; (void)out_size; (void)ws_size;
  const float* inputs = (const float*)d_in[0];
  const float* memory = (const float*)d_in[1];
  const float* Wq     = (const float*)d_in[2];
  const float* Wk     = (const float*)d_in[3];
  const float* Wv     = (const float*)d_in[4];
  const int*   mlen   = (const int*)d_in[5];
  const int*   qlen   = (const int*)d_in[6];
  const int*   causp  = (const int*)d_in[7];

  const size_t ACT = (size_t)B_ * TQ_ * 512;       // 8,388,608 elements
  const size_t WSZ = (size_t)512 * 512;            // 262,144 elements
  char* ws = (char*)d_ws;
  __bf16* Xbf = (__bf16*)ws;                            // bf16(inputs)
  __bf16* Mbf = Xbf + ACT;                              // bf16(memory)
  __bf16* WqT = Mbf + ACT;                              // Wq^T bf16
  __bf16* WkT = WqT + WSZ;
  __bf16* WvT = WkT + WSZ;
  __bf16* Qb  = WvT + WSZ;                              // Q  [B*TQ, 512]
  __bf16* Kb  = Qb + ACT;                               // K  [B*TM, 512]
  __bf16* Vt  = Kb + ACT;                               // Vt [B, 512, TM]

  float* ctx = (float*)d_out;                           // [B, TQ, 512]
  float* ali = ctx + ACT;                               // [B, H, TQ, TM]

  cvt_act<<<4096, 256, 0, stream>>>(inputs, memory, Xbf, Mbf, (int)ACT);
  cvt_wT<<<1024, 256, 0, stream>>>(Wq, Wk, Wv, WqT, WkT, WvT);

  dim3 pg((B_ * TQ_) / 16, 512 / 16, 1);                // 1024 x 32 tiles, wave per tile
  proj_gemm<<<pg, 32, 0, stream>>>(Xbf, WqT, Qb, 0);
  proj_gemm<<<pg, 32, 0, stream>>>(Mbf, WkT, Kb, 0);
  proj_gemm<<<pg, 32, 0, stream>>>(Mbf, WvT, Vt, 1);

  dim3 ag(TQ_ / 16, H_, B_);                            // 64 x 8 x 16 blocks, wave per block
  attn_kernel<<<ag, 32, 0, stream>>>(Qb, Kb, Vt, mlen, qlen, causp, ctx, ali);
}
